// NewEncoder_88064009437323
// MI455X (gfx1250) — matrix operand order repaired and checked by reading, not compiled, not run on hardware
//
#include <hip/hip_runtime.h>
#include <hip/hip_bf16.h>

// GAT 2-layer forward for gfx1250 (MI455X).
// N=50000 (multiple of 16), E=800000, IN=128, HID=256, OUT=128.

#define IN_F   128
#define HID_F  256
#define OUT_F  128
#define NEG_SLOPE 0.2f

typedef __attribute__((ext_vector_type(16))) __bf16 v16bf;
typedef __attribute__((ext_vector_type(8)))  float  v8f;

// ---------------------------------------------------------------- utilities

__global__ void zero_u32_kernel(unsigned* __restrict__ p, long long n) {
    long long i = (long long)blockIdx.x * blockDim.x + threadIdx.x;
    if (i < n) p[i] = 0u;
}

__global__ void cvt_bf16_kernel(const float* __restrict__ x, __bf16* __restrict__ y,
                                long long n) {
    long long i = (long long)blockIdx.x * blockDim.x + threadIdx.x;
    if (i < n) y[i] = (__bf16)x[i];
}

// Wt[n*K + k] = (bf16) W[k*Nf + n]   (W is [K][Nf] row-major, Wt is [Nf][K])
__global__ void transpose_cvt_kernel(const float* __restrict__ W, __bf16* __restrict__ Wt,
                                     int K, int Nf) {
    int i = blockIdx.x * blockDim.x + threadIdx.x;
    if (i < K * Nf) {
        int n = i / K, k = i % K;
        Wt[i] = (__bf16)W[(size_t)k * Nf + n];
    }
}

// ---------------------------------------------------------------- WMMA GEMM
// C(bf16)[M x Nf] = A(bf16)[M x K, row-major] * B, with B given as
// Bt(bf16)[Nf x K] (B^T, K contiguous). One wave computes a 16x64 strip
// (4 accumulators sharing one A fragment per K-step) using
// v_wmma_f32_16x16x32_bf16. Requires M%16==0, Nf%64==0, K%32==0.
__global__ void gemm_wmma_bf16(const __bf16* __restrict__ A,
                               const __bf16* __restrict__ Bt,
                               __bf16* __restrict__ C,
                               int M, int K, int Nf) {
    int wave = (int)((blockIdx.x * (long long)blockDim.x + threadIdx.x) >> 5);
    int lane = threadIdx.x & 31;
    int tilesN = Nf >> 6;          // 64-wide strips
    int tilesM = M >> 4;
    if (wave >= tilesM * tilesN) return;
    int tm = wave / tilesN;
    int tn = wave - tm * tilesN;
    int m0 = tm << 4, n0 = tn << 6;

    int row = lane & 15;           // M-row (A) / N-col (B) owned by this lane
    int hi  = lane >> 4;           // lane-group select

    // A frag per lane: row m0+row; 8 bf16 at k0+hi*8 and 8 bf16 at k0+hi*8+16
    const __bf16* arow = A  + (size_t)(m0 + row) * K + hi * 8;
    // B frag per lane (sub-tile t): col n0+16t+row; 8 bf16 at k0+hi*8 and 8 bf16 at k0+hi*8+16
    const __bf16* brow = Bt + (size_t)(n0 + row) * K + hi * 8;

    v8f c0 = {}, c1 = {}, c2 = {}, c3 = {};
    for (int k0 = 0; k0 < K; k0 += 32) {
        union { uint4 u[2]; v16bf v; } af, b0, b1, b2, b3;
        af.u[0] = *(const uint4*)(arow + k0);
        af.u[1] = *(const uint4*)(arow + k0 + 16);
        b0.u[0] = *(const uint4*)(brow + k0);
        b0.u[1] = *(const uint4*)(brow + k0 + 16);
        b1.u[0] = *(const uint4*)(brow + (size_t)16 * K + k0);
        b1.u[1] = *(const uint4*)(brow + (size_t)16 * K + k0 + 16);
        b2.u[0] = *(const uint4*)(brow + (size_t)32 * K + k0);
        b2.u[1] = *(const uint4*)(brow + (size_t)32 * K + k0 + 16);
        b3.u[0] = *(const uint4*)(brow + (size_t)48 * K + k0);
        b3.u[1] = *(const uint4*)(brow + (size_t)48 * K + k0 + 16);
        c0 = __builtin_amdgcn_wmma_f32_16x16x32_bf16(false, af.v, false, b0.v,
                                                     (short)0, c0, false, false);
        c1 = __builtin_amdgcn_wmma_f32_16x16x32_bf16(false, af.v, false, b1.v,
                                                     (short)0, c1, false, false);
        c2 = __builtin_amdgcn_wmma_f32_16x16x32_bf16(false, af.v, false, b2.v,
                                                     (short)0, c2, false, false);
        c3 = __builtin_amdgcn_wmma_f32_16x16x32_bf16(false, af.v, false, b3.v,
                                                     (short)0, c3, false, false);
    }

    // C layout: lanes 0-15 -> N=n0+16t+lane, M=m0+j; lanes 16-31 -> M=m0+8+j
    __bf16* cbase = C + (size_t)(m0 + hi * 8) * Nf + n0 + row;
    #pragma unroll
    for (int j = 0; j < 8; ++j) {
        cbase[(size_t)j * Nf +  0] = (__bf16)c0[j];
        cbase[(size_t)j * Nf + 16] = (__bf16)c1[j];
        cbase[(size_t)j * Nf + 32] = (__bf16)c2[j];
        cbase[(size_t)j * Nf + 48] = (__bf16)c3[j];
    }
}

// ---------------------------------------------------------------- attention
// alpha_src[i] = h[i].a_src ; alpha_dst[i] = h[i].a_dst   (one wave per node)
__global__ void node_scores_kernel(const __bf16* __restrict__ h,
                                   const float* __restrict__ a_src,
                                   const float* __restrict__ a_dst,
                                   float* __restrict__ as_, float* __restrict__ ad_,
                                   int n, int F) {
    long long wave = ((long long)blockIdx.x * blockDim.x + threadIdx.x) >> 5;
    int lane = threadIdx.x & 31;
    if (wave >= n) return;
    const uint2*  hrow = (const uint2*)(h + (size_t)wave * F);
    const float4* av   = (const float4*)a_src;
    const float4* dv   = (const float4*)a_dst;
    float s1 = 0.f, s2 = 0.f;
    for (int j = lane; j < (F >> 2); j += 32) {
        union { uint2 u; __bf16 b[4]; } q;
        q.u = hrow[j];
        float4 a = av[j], d = dv[j];
        float h0 = (float)q.b[0], h1 = (float)q.b[1];
        float h2 = (float)q.b[2], h3 = (float)q.b[3];
        s1 += h0 * a.x + h1 * a.y + h2 * a.z + h3 * a.w;
        s2 += h0 * d.x + h1 * d.y + h2 * d.z + h3 * d.w;
    }
    #pragma unroll
    for (int off = 16; off; off >>= 1) {
        s1 += __shfl_xor(s1, off, 32);
        s2 += __shfl_xor(s2, off, 32);
    }
    if (lane == 0) { as_[wave] = s1; ad_[wave] = s2; }
}

__device__ __forceinline__ unsigned f32_key(float f) {
    int b = __float_as_int(f);
    return (b >= 0) ? ((unsigned)b | 0x80000000u) : ~(unsigned)b;
}
__device__ __forceinline__ float f32_unkey(unsigned u) {
    int b = (u & 0x80000000u) ? (int)(u ^ 0x80000000u) : ~(int)u;
    return __int_as_float(b);
}

// logit per edge + segment max via monotonic-u32 atomicMax
__global__ void edge_logits_kernel(const float* __restrict__ as_,
                                   const float* __restrict__ ad_,
                                   const int* __restrict__ src,
                                   const int* __restrict__ dst,
                                   long long E, long long N,
                                   float* __restrict__ elog,
                                   unsigned* __restrict__ nmax) {
    long long t = (long long)blockIdx.x * blockDim.x + threadIdx.x;
    if (t >= E + N) return;
    int s = (t < E) ? src[t] : (int)(t - E);
    int d = (t < E) ? dst[t] : (int)(t - E);
    float x = as_[s] + ad_[d];
    float e = (x > 0.f) ? x : NEG_SLOPE * x;
    elog[t] = e;
    atomicMax(&nmax[d], f32_key(e));
}

// p = exp(e - max[dst]); denom[dst] += p
__global__ void edge_p_kernel(float* __restrict__ elog,
                              const unsigned* __restrict__ nmax,
                              const int* __restrict__ dst,
                              long long E, long long N,
                              float* __restrict__ denom) {
    long long t = (long long)blockIdx.x * blockDim.x + threadIdx.x;
    if (t >= E + N) return;
    int d = (t < E) ? dst[t] : (int)(t - E);
    float p = __expf(elog[t] - f32_unkey(nmax[d]));
    elog[t] = p;
    atomicAdd(&denom[d], p);
}

// acc[dst] += p * h[src]  (one wave per edge; h/acc rows are L2-resident)
__global__ void edge_scatter_kernel(const __bf16* __restrict__ h,
                                    const float* __restrict__ p,
                                    const int* __restrict__ src,
                                    const int* __restrict__ dst,
                                    long long E, long long N,
                                    float* __restrict__ acc, int F) {
    long long wave = ((long long)blockIdx.x * blockDim.x + threadIdx.x) >> 5;
    int lane = threadIdx.x & 31;
    if (wave >= E + N) return;
    int s = (wave < E) ? src[wave] : (int)(wave - E);
    int d = (wave < E) ? dst[wave] : (int)(wave - E);
    float pv = p[wave];
    const uint2* hr = (const uint2*)(h + (size_t)s * F);   // 4 bf16 per load
    float* ar = acc + (size_t)d * F;
    __builtin_prefetch(hr + lane, 0, 0);                   // global_prefetch_b8
    for (int j = lane; j < (F >> 2); j += 32) {
        union { uint2 u; __bf16 b[4]; } q;
        q.u = hr[j];
        int f = j << 2;
        atomicAdd(&ar[f + 0], pv * (float)q.b[0]);
        atomicAdd(&ar[f + 1], pv * (float)q.b[1]);
        atomicAdd(&ar[f + 2], pv * (float)q.b[2]);
        atomicAdd(&ar[f + 3], pv * (float)q.b[3]);
    }
}

// out = relu(acc/denom + b); optional f32 and/or bf16 outputs
__global__ void finalize_kernel(const float* __restrict__ acc,
                                const float* __restrict__ denom,
                                const float* __restrict__ bias,
                                long long n, int F,
                                float* __restrict__ outf,
                                __bf16* __restrict__ outb) {
    long long i = (long long)blockIdx.x * blockDim.x + threadIdx.x;
    if (i >= n * F) return;
    long long node = i / F;
    int f = (int)(i - node * F);
    float v = acc[i] / (denom[node] + 1e-16f) + bias[f];
    v = (v > 0.f) ? v : 0.f;
    if (outf) outf[i] = v;
    if (outb) outb[i] = (__bf16)v;
}

// ---------------------------------------------------------------- launch

static inline unsigned cdiv_u(long long a, long long b) {
    return (unsigned)((a + b - 1) / b);
}

extern "C" void kernel_launch(void* const* d_in, const int* in_sizes, int n_in,
                              void* d_out, int out_size, void* d_ws, size_t ws_size,
                              hipStream_t stream) {
    const float* x   = (const float*)d_in[0];
    const int*   ei  = (const int*)  d_in[1];
    const float* W1  = (const float*)d_in[2];
    const float* a1s = (const float*)d_in[3];
    const float* a1d = (const float*)d_in[4];
    const float* b1  = (const float*)d_in[5];
    const float* W2  = (const float*)d_in[6];
    const float* a2s = (const float*)d_in[7];
    const float* a2d = (const float*)d_in[8];
    const float* b2  = (const float*)d_in[9];
    float* out = (float*)d_out;

    long long N  = in_sizes[0] / IN_F;
    long long E  = in_sizes[1] / 2;
    long long ET = E + N;
    const int* src = ei;
    const int* dst = ei + E;

    // ---- workspace carve-out (256B aligned slabs)
    char* w = (char*)d_ws;
    auto carve = [&](size_t bytes) -> char* {
        char* p = w;
        w += (bytes + 255) & ~(size_t)255;
        return p;
    };
    __bf16*  xb    = (__bf16*)carve((size_t)N * IN_F * 2);     // x in bf16
    __bf16*  w1t   = (__bf16*)carve((size_t)IN_F * HID_F * 2); // W1^T bf16
    __bf16*  w2t   = (__bf16*)carve((size_t)HID_F * OUT_F * 2);// W2^T bf16
    __bf16*  h     = (__bf16*)carve((size_t)N * HID_F * 2);    // GEMM out (both layers)
    __bf16*  h2in  = (__bf16*)carve((size_t)N * HID_F * 2);    // layer-1 final (bf16)
    float*   acc   = (float*) carve((size_t)N * HID_F * 4);    // f32 accumulators
    float*   as_   = (float*) carve((size_t)N * 4);
    float*   ad_   = (float*) carve((size_t)N * 4);
    unsigned* nmax = (unsigned*)carve((size_t)N * 4);
    float*   denom = (float*) carve((size_t)N * 4);
    float*   ep    = (float*) carve((size_t)ET * 4);
    (void)ws_size; (void)n_in; (void)out_size;

    const int TB = 256;

    // ---- one-time conversions
    cvt_bf16_kernel<<<cdiv_u(N * IN_F, TB), TB, 0, stream>>>(x, xb, N * IN_F);
    transpose_cvt_kernel<<<cdiv_u((long long)IN_F * HID_F, TB), TB, 0, stream>>>(W1, w1t, IN_F, HID_F);
    transpose_cvt_kernel<<<cdiv_u((long long)HID_F * OUT_F, TB), TB, 0, stream>>>(W2, w2t, HID_F, OUT_F);

    // ================= layer 1:  IN_F -> HID_F =================
    {
        long long waves = (N >> 4) * (HID_F >> 6);           // 16x64 strips
        gemm_wmma_bf16<<<cdiv_u(waves * 32, TB), TB, 0, stream>>>(
            xb, w1t, h, (int)N, IN_F, HID_F);

        node_scores_kernel<<<cdiv_u(N * 32, TB), TB, 0, stream>>>(
            h, a1s, a1d, as_, ad_, (int)N, HID_F);

        zero_u32_kernel<<<cdiv_u(N * HID_F, TB), TB, 0, stream>>>((unsigned*)acc, N * HID_F);
        zero_u32_kernel<<<cdiv_u(N, TB), TB, 0, stream>>>(nmax, N);
        zero_u32_kernel<<<cdiv_u(N, TB), TB, 0, stream>>>((unsigned*)denom, N);

        edge_logits_kernel<<<cdiv_u(ET, TB), TB, 0, stream>>>(as_, ad_, src, dst, E, N, ep, nmax);
        edge_p_kernel<<<cdiv_u(ET, TB), TB, 0, stream>>>(ep, nmax, dst, E, N, denom);
        edge_scatter_kernel<<<cdiv_u(ET * 32, TB), TB, 0, stream>>>(
            h, ep, src, dst, E, N, acc, HID_F);

        finalize_kernel<<<cdiv_u(N * HID_F, TB), TB, 0, stream>>>(
            acc, denom, b1, N, HID_F, (float*)nullptr, h2in);
    }

    // ================= layer 2:  HID_F -> OUT_F =================
    {
        long long waves = (N >> 4) * (OUT_F >> 6);
        gemm_wmma_bf16<<<cdiv_u(waves * 32, TB), TB, 0, stream>>>(
            h2in, w2t, h, (int)N, HID_F, OUT_F);

        node_scores_kernel<<<cdiv_u(N * 32, TB), TB, 0, stream>>>(
            h, a2s, a2d, as_, ad_, (int)N, OUT_F);

        zero_u32_kernel<<<cdiv_u(N * OUT_F, TB), TB, 0, stream>>>((unsigned*)acc, N * OUT_F);
        zero_u32_kernel<<<cdiv_u(N, TB), TB, 0, stream>>>(nmax, N);
        zero_u32_kernel<<<cdiv_u(N, TB), TB, 0, stream>>>((unsigned*)denom, N);

        edge_logits_kernel<<<cdiv_u(ET, TB), TB, 0, stream>>>(as_, ad_, src, dst, E, N, ep, nmax);
        edge_p_kernel<<<cdiv_u(ET, TB), TB, 0, stream>>>(ep, nmax, dst, E, N, denom);
        edge_scatter_kernel<<<cdiv_u(ET * 32, TB), TB, 0, stream>>>(
            h, ep, src, dst, E, N, acc, OUT_F);

        finalize_kernel<<<cdiv_u(N * OUT_F, TB), TB, 0, stream>>>(
            acc, denom, b2, N, OUT_F, out, (__bf16*)nullptr);
    }
}
